// HMM_43765716746857
// MI455X (gfx1250) — matrix operand order repaired and checked by reading, not compiled, not run on hardware
//
#include <hip/hip_runtime.h>

typedef __attribute__((ext_vector_type(16))) _Float16 v16h;
typedef __attribute__((ext_vector_type(8)))  _Float16 v8h;
typedef __attribute__((ext_vector_type(8)))  float    v8f;
typedef __attribute__((ext_vector_type(4)))  float    v4f;

constexpr int HB = 64;     // batch
constexpr int HT = 512;    // time steps
constexpr int HS = 512;    // states
constexpr int HV = 2048;   // vocab
constexpr int PSTR = 520;  // padded LDS row stride in halfs (1040B: 16B-aligned, bank-skewed)

// ---------------------------------------------------------------------------
// Preprocessing: expTt[j][i] = softmax(trans[i,:])[j]  (transposed, f16)
// ---------------------------------------------------------------------------
__global__ __launch_bounds__(256) void hmm_prep_trans(const float* __restrict__ trans,
                                                      _Float16* __restrict__ expTt) {
    const int i = blockIdx.x;      // source state (row of trans)
    const int tid = threadIdx.x;
    __shared__ float red[256];
    float x0 = trans[i * HS + tid];
    float x1 = trans[i * HS + 256 + tid];
    red[tid] = fmaxf(x0, x1);
    __syncthreads();
    for (int s = 128; s > 0; s >>= 1) {
        if (tid < s) red[tid] = fmaxf(red[tid], red[tid + s]);
        __syncthreads();
    }
    float m = red[0];
    __syncthreads();
    float e0 = __expf(x0 - m), e1 = __expf(x1 - m);
    red[tid] = e0 + e1;
    __syncthreads();
    for (int s = 128; s > 0; s >>= 1) {
        if (tid < s) red[tid] += red[tid + s];
        __syncthreads();
    }
    float inv = 1.0f / red[0];
    expTt[tid * HS + i]         = (_Float16)(e0 * inv);   // j = tid
    expTt[(tid + 256) * HS + i] = (_Float16)(e1 * inv);   // j = tid+256
}

// ---------------------------------------------------------------------------
// Preprocessing: emitLse[s] = logsumexp_v emit[s,:]; block HS does log-prior
// ---------------------------------------------------------------------------
__global__ __launch_bounds__(256) void hmm_prep_emit(const float* __restrict__ emitL,
                                                     const float* __restrict__ priorL,
                                                     float* __restrict__ emitLse,
                                                     float* __restrict__ logPrior) {
    const int tid = threadIdx.x;
    __shared__ float red[256];
    if (blockIdx.x < HS) {
        const float* row = emitL + (size_t)blockIdx.x * HV;
        float mx = -INFINITY;
        for (int e = tid; e < HV; e += 256) mx = fmaxf(mx, row[e]);
        red[tid] = mx;
        __syncthreads();
        for (int s = 128; s > 0; s >>= 1) {
            if (tid < s) red[tid] = fmaxf(red[tid], red[tid + s]);
            __syncthreads();
        }
        float m = red[0];
        __syncthreads();
        float sum = 0.0f;
        for (int e = tid; e < HV; e += 256) sum += __expf(row[e] - m);
        red[tid] = sum;
        __syncthreads();
        for (int s = 128; s > 0; s >>= 1) {
            if (tid < s) red[tid] += red[tid + s];
            __syncthreads();
        }
        if (tid == 0) emitLse[blockIdx.x] = m + __logf(red[0]);
    } else {
        float x0 = priorL[tid], x1 = priorL[tid + 256];
        red[tid] = fmaxf(x0, x1);
        __syncthreads();
        for (int s = 128; s > 0; s >>= 1) {
            if (tid < s) red[tid] = fmaxf(red[tid], red[tid + s]);
            __syncthreads();
        }
        float m = red[0];
        __syncthreads();
        red[tid] = __expf(x0 - m) + __expf(x1 - m);
        __syncthreads();
        for (int s = 128; s > 0; s >>= 1) {
            if (tid < s) red[tid] += red[tid + s];
            __syncthreads();
        }
        float lse = m + __logf(red[0]);
        logPrior[tid]       = x0 - lse;
        logPrior[tid + 256] = x1 - lse;
    }
}

// ---------------------------------------------------------------------------
// alpha0[b,s] = em(0,b,s) + logPrior[s]; also resets the grid barrier state
// ---------------------------------------------------------------------------
__global__ __launch_bounds__(256) void hmm_init_alpha(const float* __restrict__ emitL,
                                                      const int* __restrict__ value,
                                                      const float* __restrict__ emitLse,
                                                      const float* __restrict__ logPrior,
                                                      float* __restrict__ alpha,
                                                      int* __restrict__ bar) {
    const int idx = blockIdx.x * 256 + threadIdx.x;   // 0..HB*HS-1
    const int b = idx >> 9, s = idx & (HS - 1);
    const int obs = value[b * HT];                     // t = 0
    alpha[idx] = emitL[(size_t)s * HV + obs] - emitLse[s] + logPrior[s];
    if (idx < 2) bar[idx] = 0;                         // deterministic reset every call
}

// ---------------------------------------------------------------------------
// Device-wide sense-reversal barrier (16 blocks, all resident)
// ---------------------------------------------------------------------------
__device__ __forceinline__ void gridBarrier(int* cnt, int* gen, int nb) {
    __threadfence();     // every wave: own stores visible at device scope
    __syncthreads();
    if (threadIdx.x == 0) {
        int g = __hip_atomic_load(gen, __ATOMIC_RELAXED, __HIP_MEMORY_SCOPE_AGENT);
        int prev = __hip_atomic_fetch_add(cnt, 1, __ATOMIC_ACQ_REL, __HIP_MEMORY_SCOPE_AGENT);
        if (prev == nb - 1) {
            __hip_atomic_store(cnt, 0, __ATOMIC_RELAXED, __HIP_MEMORY_SCOPE_AGENT);
            __hip_atomic_fetch_add(gen, 1, __ATOMIC_RELEASE, __HIP_MEMORY_SCOPE_AGENT);
        } else {
            while (__hip_atomic_load(gen, __ATOMIC_ACQUIRE, __HIP_MEMORY_SCOPE_AGENT) == g)
                __builtin_amdgcn_s_sleep(1);
        }
    }
    __syncthreads();
}

// ---------------------------------------------------------------------------
// Persistent forward kernel: 16 blocks x 256 threads = 128 waves = 128 tiles
//   block: row-tile rt = blk&3 (16 batch rows), col-group ctg = blk>>2
//   wave:  col-tile ct = ctg*8 + waveId  (16 next-states)
// ---------------------------------------------------------------------------
__global__ __launch_bounds__(256) void hmm_forward(const float* __restrict__ emitL,
                                                   const int* __restrict__ value,
                                                   const _Float16* __restrict__ expTt,
                                                   const float* __restrict__ emitLse,
                                                   float* __restrict__ alpha,
                                                   int* __restrict__ bar,
                                                   float* __restrict__ out) {
    const int tid  = threadIdx.x;
    const int lane = tid & 31;
    const int wid  = tid >> 5;              // 0..7
    const int rt   = blockIdx.x & 3;        // row tile (batch)
    const int ctg  = blockIdx.x >> 2;       // 0..3
    const int ct   = ctg * 8 + wid;         // column tile (state) 0..31
    const int rowBase = rt * 16;

    __shared__ __align__(32) _Float16 lP[16][PSTR];  // exp-normalized alpha rows
    __shared__ float lm[16];                         // per-row max
    __shared__ int   lobs[16];                       // observations at step t
    __shared__ float lLse[128];                      // emitLse for this block's columns
    __shared__ float red[16][17];

    if (tid < 128) lLse[tid] = emitLse[ctg * 128 + tid];

    // --- B fragments (expT columns for this wave) loaded ONCE into registers.
    // B layout used here (K=32, f16): lanes 0-15 hold K=0..15 contiguous, lanes 16-31 hold K=16..31.
    const int n  = lane & 15;
    const int hi = lane >> 4;
    const int j  = ct * 16 + n;             // global next-state column
    v16h bFrag[16];
#pragma unroll
    for (int c = 0; c < 16; ++c)
        bFrag[c] = *(const v16h*)(expTt + (size_t)j * HS + c * 32 + hi * 16);

    const int r = tid >> 4;                 // local batch row 0..15 (phase 1)
    const int q = tid & 15;                 // 32-element chunk within row

    for (int t = 1; t < HT; ++t) {
        const float* aP = alpha + ((t + 1) & 1) * (HB * HS);
        float*       aN = alpha + (t & 1) * (HB * HS);

        // ---- phase 1: rows rowBase..rowBase+15 -> max, P = exp(alpha-m) in LDS
        const float* rowp = aP + (rowBase + r) * HS + q * 32;
        float mx = -INFINITY;
#pragma unroll
        for (int e = 0; e < 32; e += 4) {
            v4f v = *(const v4f*)(rowp + e);
            mx = fmaxf(mx, fmaxf(fmaxf(v.x, v.y), fmaxf(v.z, v.w)));
        }
        red[r][q] = mx;
        __syncthreads();
        if (q == 0) {
            float m = red[r][0];
#pragma unroll
            for (int e = 1; e < 16; ++e) m = fmaxf(m, red[r][e]);
            lm[r] = m;
        }
        if (tid < 16) lobs[tid] = value[(rowBase + tid) * HT + t];
        __syncthreads();
        {
            float m = lm[r];
            _Float16* dst = &lP[r][q * 32];
#pragma unroll
            for (int e = 0; e < 32; ++e) dst[e] = (_Float16)__expf(rowp[e] - m);
        }
        __syncthreads();

        // ---- phase 2: 16x16 tile GEMM over K=512 via v_wmma_f32_16x16x32_f16
        // A layout (f16 16x32, matches B): lanes 0-15: e0-15 = K0-15;
        //                                   lanes16-31: e0-15 = K16-31.
        v8f acc = {0.f, 0.f, 0.f, 0.f, 0.f, 0.f, 0.f, 0.f};
        const _Float16* pr = &lP[n][0];     // A row m = lane&15
#pragma unroll
        for (int c = 0; c < 16; ++c) {
            const int k0 = c * 32 + hi * 16;
            v8h lo = *(const v8h*)(pr + k0);
            v8h hp = *(const v8h*)(pr + k0 + 8);
            v16h a = __builtin_shufflevector(lo, hp, 0, 1, 2, 3, 4, 5, 6, 7,
                                                     8, 9, 10, 11, 12, 13, 14, 15);
            acc = __builtin_amdgcn_wmma_f32_16x16x32_f16(
                false, a, false, bFrag[c], (short)0, acc, false, false);
        }

        // ---- epilogue: alpha'[b,j] = em + m_b + log(dot)
        // C layout: VGPR v, lanes 0-15 -> M=v, lanes 16-31 -> M=v+8; N = lane&15
        const float lseJ = lLse[wid * 16 + n];
#pragma unroll
        for (int v = 0; v < 8; ++v) {
            const int rl   = v + hi * 8;             // local batch row
            const int brow = rowBase + rl;
            const float em = emitL[(size_t)j * HV + lobs[rl]] - lseJ;
            aN[brow * HS + j] = em + lm[rl] + __logf(acc[v]);
        }

        gridBarrier(bar, bar + 1, (int)gridDim.x);
    }

    // ---- final: out[b] = logsumexp_s alpha_final[b,s]   (block 0 only)
    if (blockIdx.x == 0) {
        const float* aF = alpha + ((HT - 1) & 1) * (HB * HS);
        const int fr = tid >> 2, fq = tid & 3;       // 4 threads per batch row
        const float* rowf = aF + fr * HS + fq * 128;
        float mx = -INFINITY;
        for (int e = 0; e < 128; e += 4) {
            v4f v = *(const v4f*)(rowf + e);
            mx = fmaxf(mx, fmaxf(fmaxf(v.x, v.y), fmaxf(v.z, v.w)));
        }
        mx = fmaxf(mx, __shfl_xor(mx, 1));
        mx = fmaxf(mx, __shfl_xor(mx, 2));
        float sum = 0.0f;
        for (int e = 0; e < 128; ++e) sum += __expf(rowf[e] - mx);
        sum += __shfl_xor(sum, 1);
        sum += __shfl_xor(sum, 2);
        if (fq == 0) out[fr] = mx + __logf(sum);
    }
}

// ---------------------------------------------------------------------------
extern "C" void kernel_launch(void* const* d_in, const int* in_sizes, int n_in,
                              void* d_out, int out_size, void* d_ws, size_t ws_size,
                              hipStream_t stream) {
    const int*   value  = (const int*)  d_in[0];   // [B,T] int32
    const float* priorL = (const float*)d_in[1];   // [S]
    const float* transL = (const float*)d_in[2];   // [S,S]
    const float* emitL  = (const float*)d_in[3];   // [S,V]
    float* out = (float*)d_out;                    // [B]

    char* ws = (char*)d_ws;
    _Float16* expTt    = (_Float16*)ws;                               // 512 KB
    float*    alpha    = (float*)(ws + 524288);                       // 256 KB (double buffer)
    float*    emitLse  = (float*)(ws + 524288 + 262144);              // 2 KB
    float*    logPrior = (float*)(ws + 524288 + 262144 + 2048);       // 2 KB
    int*      bar      = (int*)  (ws + 524288 + 262144 + 4096);       // 8 B

    hmm_prep_trans<<<HS, 256, 0, stream>>>(transL, expTt);
    hmm_prep_emit <<<HS + 1, 256, 0, stream>>>(emitL, priorL, emitLse, logPrior);
    hmm_init_alpha<<<(HB * HS) / 256, 256, 0, stream>>>(emitL, value, emitLse, logPrior,
                                                        alpha, bar);
    hmm_forward   <<<16, 256, 0, stream>>>(emitL, value, expTt, emitLse, alpha, bar, out);
}
